// MultiHeadAttention_readout_41137196761650
// MI455X (gfx1250) — matrix operand order repaired and checked by reading, not compiled, not run on hardware
//
#include <hip/hip_runtime.h>
#include <hip/hip_bf16.h>

typedef __attribute__((ext_vector_type(16))) _Float16 v16h;
typedef __attribute__((ext_vector_type(8)))  float    v8f;
typedef __attribute__((ext_vector_type(4)))  int      v4i;

#define B_    256
#define D_    256
#define H_    8
#define DH_   32
#define LP1   257
#define LPAD  272   // 17 * 16, padded sequence length for tiling
#define MT    17    // M tiles per sample

// ---- gfx1250 async global->LDS path (guarded; falls back to register copies) ----
#if defined(__has_builtin)
#if __has_builtin(__builtin_amdgcn_global_load_async_to_lds_b128) && \
    __has_builtin(__builtin_amdgcn_s_wait_asynccnt)
#define HAVE_ASYNC_LDS 1
#endif
#endif
#ifndef HAVE_ASYNC_LDS
#define HAVE_ASYNC_LDS 0
#endif

#if HAVE_ASYNC_LDS
typedef __attribute__((address_space(1))) v4i gv4i;  // global int4
typedef __attribute__((address_space(3))) v4i lv4i;  // LDS int4
#endif

// copy 16 bytes global -> LDS (async when available)
static __device__ __forceinline__ void copy16_g2lds(_Float16* lds, const _Float16* g) {
#if HAVE_ASYNC_LDS
  __builtin_amdgcn_global_load_async_to_lds_b128((gv4i*)g, (lv4i*)lds, 0, 0);
#else
  *(int4*)lds = *(const int4*)g;
#endif
}

static __device__ __forceinline__ void wait_async_copies() {
#if HAVE_ASYNC_LDS
  __builtin_amdgcn_s_wait_asynccnt(0);
#endif
}

// ---------------- WMMA fragment loaders (CDNA5 16x16x32 f16 layouts) ----------------
// A (16x32 MxK): lane&15 = M; lanes<16 hold K {0..15}, lanes>=16 hold K {16..31}
static __device__ __forceinline__ v16h load_a_frag(const _Float16* A, int ldA, int row0, int col0) {
  const int lane = threadIdx.x & 31;
  const _Float16* p = A + (size_t)(row0 + (lane & 15)) * ldA + col0 + ((lane & 16) ? 16 : 0);
  v16h a;
#pragma unroll
  for (int i = 0; i < 16; ++i) a[i] = p[i];
  return a;
}

// B operand (32x16 KxN) read from a row-major [N x K] matrix (i.e. transposed storage):
// element B[k][n] = M[n0+n][k0+k]; per-lane reads 16 contiguous halves -> 2x b128.
static __device__ __forceinline__ v16h load_bT_frag(const _Float16* M, int ldM, int n0, int k0) {
  const int lane = threadIdx.x & 31;
  const _Float16* p = M + (size_t)(n0 + (lane & 15)) * ldM + k0 + ((lane & 16) ? 16 : 0);
  v16h b;
#pragma unroll
  for (int i = 0; i < 16; ++i) b[i] = p[i];
  return b;
}

// ---------------- Kernel 1: scatter packed tokens into padded f16 buffer ----------------
__global__ void prep_pad(const float* __restrict__ x, const int* __restrict__ lengths,
                         const float* __restrict__ cls_emb, _Float16* __restrict__ P) {
  const int b = blockIdx.x;
  const int t = threadIdx.x;  // column d
  __shared__ int s_start, s_len;
  if (t == 0) {
    int acc = 0;
    for (int i = 0; i < b; ++i) acc += lengths[i];
    s_start = acc;
    s_len = lengths[b];
  }
  __syncthreads();
  const int start = s_start, len = s_len;
  _Float16* Pb = P + (size_t)b * LPAD * D_;
  for (int r = 0; r < LPAD; ++r) {
    float v;
    if (r < len)       v = x[(size_t)(start + r) * D_ + t];
    else if (r == len) v = cls_emb[t];
    else               v = 0.0f;
    Pb[(size_t)r * D_ + t] = (_Float16)v;
  }
}

// ---------------- Kernel 2: f32 -> f16 weight conversion WITH transpose ----------------
// Produces W^T stored row-major [N x K] so B-fragment loads are contiguous per lane.
__global__ void conv_w(const float* __restrict__ Wq, const float* __restrict__ Wk,
                       const float* __restrict__ Wv, const float* __restrict__ Wo,
                       _Float16* __restrict__ WqT, _Float16* __restrict__ WkT,
                       _Float16* __restrict__ WvT, _Float16* __restrict__ WoT) {
  const int k = blockIdx.x;    // row of original W
  const int n = threadIdx.x;   // column of original W
  const size_t src = (size_t)k * D_ + n;
  const size_t dst = (size_t)n * D_ + k;
  WqT[dst] = (_Float16)Wq[src];
  WkT[dst] = (_Float16)Wk[src];
  WvT[dst] = (_Float16)Wv[src];
  WoT[dst] = (_Float16)Wo[src];
}

// ---------------- Kernel 3: Q/K/V projection GEMMs via WMMA ----------------
// grid = (17 Mtiles, 3 matrices, B). 8 waves/block, wave w does N-tiles {w, w+8}.
__global__ void qkv_gemm(const _Float16* __restrict__ P,
                         const _Float16* __restrict__ WqT, const _Float16* __restrict__ WkT,
                         const _Float16* __restrict__ WvT,
                         _Float16* __restrict__ Qh, _Float16* __restrict__ Kh,
                         _Float16* __restrict__ Vh) {
  const int mt   = blockIdx.x;
  const int mat  = blockIdx.y;
  const int b    = blockIdx.z;
  const int wave = threadIdx.x >> 5;
  const int lane = threadIdx.x & 31;

  const _Float16* WT = (mat == 0) ? WqT : (mat == 1) ? WkT : WvT;
  _Float16*      Out = (mat == 0) ? Qh  : (mat == 1) ? Kh  : Vh;

  const _Float16* Ab = P + (size_t)b * LPAD * D_;
  const int row0 = mt * 16;
  const int n0 = wave * 16;
  const int n1 = (wave + 8) * 16;

  v8f c0 = {}; v8f c1 = {};
#pragma unroll 1
  for (int kt = 0; kt < 8; ++kt) {
    if (kt < 7) __builtin_prefetch(WT + (size_t)n0 * D_ + (kt + 1) * 32, 0, 0);
    v16h a  = load_a_frag(Ab, D_, row0, kt * 32);
    v16h b0 = load_bT_frag(WT, D_, n0, kt * 32);
    v16h b1 = load_bT_frag(WT, D_, n1, kt * 32);
    c0 = __builtin_amdgcn_wmma_f32_16x16x32_f16(false, a, false, b0, (short)0, c0, false, false);
    c1 = __builtin_amdgcn_wmma_f32_16x16x32_f16(false, a, false, b1, (short)0, c1, false, false);
  }

  const float scale = (mat == 0) ? 0.17677669529663687f : 1.0f;  // fold 1/sqrt(DH) into Q
  _Float16* Ob = Out + (size_t)b * LPAD * D_;
  const int rbase = row0 + ((lane & 16) ? 8 : 0);
  const int cA = n0 + (lane & 15);
  const int cB = n1 + (lane & 15);
#pragma unroll
  for (int r = 0; r < 8; ++r) {
    Ob[(size_t)(rbase + r) * D_ + cA] = (_Float16)(c0[r] * scale);
    Ob[(size_t)(rbase + r) * D_ + cB] = (_Float16)(c1[r] * scale);
  }
}

// ---------------- Kernel 4: scores + masked softmax + alpha write + CLS-row * V ----------------
// grid = (17 Mtiles, H, B). Block = 256 threads (8 waves).
// Q tile (16x32) and K head-slice (272x32) are staged into LDS with async b128 copies,
// then all WMMA operands come from LDS (ds_load_b128).
__global__ void attn_softmax(const _Float16* __restrict__ Qh, const _Float16* __restrict__ Kh,
                             const _Float16* __restrict__ Vh, const int* __restrict__ lengths,
                             float* __restrict__ alpha, _Float16* __restrict__ HV16) {
  const int mt = blockIdx.x;
  const int h  = blockIdx.y;
  const int b  = blockIdx.z;
  const int t  = threadIdx.x;
  const int wave = t >> 5, lane = t & 31;
  const int len = lengths[b];
  const int m0  = mt * 16;

  __shared__ _Float16 Ks[LPAD][DH_];   // 17,408 B
  __shared__ _Float16 Qs[16][DH_];     //  1,024 B
  __shared__ float E[16][273];
  __shared__ float red[16][16];
  __shared__ float rmax[16], rsum[16];
  __shared__ float s_hv[8][32];

  const _Float16* Qb = Qh + (size_t)b * LPAD * D_ + h * DH_;
  const _Float16* Kb = Kh + (size_t)b * LPAD * D_ + h * DH_;

  // Stage K slice: 272 rows x 64B = 1088 16B-chunks; 4 chunks/row.
  for (int chunk = t; chunk < LPAD * 4; chunk += 256) {
    const int row = chunk >> 2;
    const int co  = (chunk & 3) * 8;   // 8 halves per 16B chunk
    copy16_g2lds(&Ks[row][co], Kb + (size_t)row * D_ + co);
  }
  // Stage Q tile: 16 rows x 64B = 64 chunks (threads 0..63).
  for (int chunk = t; chunk < 16 * 4; chunk += 256) {
    const int row = chunk >> 2;
    const int co  = (chunk & 3) * 8;
    copy16_g2lds(&Qs[row][co], Qb + (size_t)(m0 + row) * D_ + co);
  }
  wait_async_copies();
  __syncthreads();

  // One WMMA per 16x16 score tile: K-dim = DH = 32 exactly. Operands from LDS.
  v16h a = load_a_frag(&Qs[0][0], DH_, 0, 0);
  for (int ct = wave; ct < MT; ct += 8) {
    v16h bb = load_bT_frag(&Ks[0][0], DH_, ct * 16, 0);
    v8f c = {};
    c = __builtin_amdgcn_wmma_f32_16x16x32_f16(false, a, false, bb, (short)0, c, false, false);
    const int col = ct * 16 + (lane & 15);
    const int rb  = (lane & 16) ? 8 : 0;
#pragma unroll
    for (int r = 0; r < 8; ++r) {
      const int gm = m0 + rb + r;
      E[rb + r][col] = ((gm <= len) && (col <= len)) ? c[r] : -1.0e10f;
    }
  }
  __syncthreads();

  // row max over cols [0, 257)
  {
    const int r = t >> 4, j = t & 15;
    float mx = -3.0e38f;
    for (int c = j; c < LP1; c += 16) mx = fmaxf(mx, E[r][c]);
    red[r][j] = mx;
  }
  __syncthreads();
  if (t < 16) {
    float mx = red[t][0];
    for (int j = 1; j < 16; ++j) mx = fmaxf(mx, red[t][j]);
    rmax[t] = mx;
  }
  __syncthreads();
  // row sum of exp
  {
    const int r = t >> 4, j = t & 15;
    float sm = 0.0f;
    for (int c = j; c < LP1; c += 16) sm += __expf(E[r][c] - rmax[r]);
    red[r][j] = sm;
  }
  __syncthreads();
  if (t < 16) {
    float sm = 0.0f;
    for (int j = 0; j < 16; ++j) sm += red[t][j];
    rsum[t] = sm;
  }
  __syncthreads();

  // normalize in LDS and stream alpha to global (coalesced f32 rows)
  float* Ao = alpha + (size_t)(b * H_ + h) * LP1 * LP1;
#pragma unroll 1
  for (int r = 0; r < 16; ++r) {
    const int gm = m0 + r;
    if (gm >= LP1) break;
    const float mx = rmax[r];
    const float inv = 1.0f / rsum[r];
    for (int c = t; c < LP1; c += 256) {
      const float v = __expf(E[r][c] - mx) * inv;
      E[r][c] = v;
      Ao[(size_t)gm * LP1 + c] = v;
    }
  }
  __syncthreads();

  // If this tile owns the CLS row (gm == len), compute hv[b,h,:] = alpha_row . V
  if (len >= m0 && len < m0 + 16) {
    const int rr = len - m0;
    const int d = t & 31, part = t >> 5;
    const _Float16* Vb = Vh + (size_t)b * LPAD * D_ + h * DH_ + d;
    float acc = 0.0f;
    for (int y = part; y < LP1; y += 8)
      acc += E[rr][y] * (float)Vb[(size_t)y * D_];
    s_hv[part][d] = acc;
    __syncthreads();
    if (t < 32) {
      float s = 0.0f;
#pragma unroll
      for (int p = 0; p < 8; ++p) s += s_hv[p][t];
      HV16[(size_t)b * D_ + h * DH_ + t] = (_Float16)s;
    }
  }
}

// ---------------- Kernel 5: cls = hv @ Wo via WMMA (256x256x256) ----------------
// 32 blocks * 8 waves = 256 waves; one 16x16 tile per wave.
__global__ void cls_out(const _Float16* __restrict__ HV16, const _Float16* __restrict__ WoT,
                        float* __restrict__ cls) {
  const int wave = threadIdx.x >> 5, lane = threadIdx.x & 31;
  const int tile = blockIdx.x * 8 + wave;   // 0..255
  const int mt = tile >> 4, nt = tile & 15;
  v8f c = {};
#pragma unroll 1
  for (int kt = 0; kt < 8; ++kt) {
    v16h a  = load_a_frag(HV16, D_, mt * 16, kt * 32);
    v16h bb = load_bT_frag(WoT, D_, nt * 16, kt * 32);
    c = __builtin_amdgcn_wmma_f32_16x16x32_f16(false, a, false, bb, (short)0, c, false, false);
  }
  const int rbase = mt * 16 + ((lane & 16) ? 8 : 0);
  const int col = nt * 16 + (lane & 15);
#pragma unroll
  for (int r = 0; r < 8; ++r)
    cls[(size_t)(rbase + r) * D_ + col] = c[r];
}

// ---------------- Launch ----------------
extern "C" void kernel_launch(void* const* d_in, const int* in_sizes, int n_in,
                              void* d_out, int out_size, void* d_ws, size_t ws_size,
                              hipStream_t stream) {
  const float* x       = (const float*)d_in[0];
  const int*   lengths = (const int*)  d_in[1];
  // d_in[2] = max_len (static 256, unused at runtime)
  const float* Wq      = (const float*)d_in[3];
  const float* Wk      = (const float*)d_in[4];
  const float* Wv      = (const float*)d_in[5];
  const float* Wo      = (const float*)d_in[6];
  const float* cls_emb = (const float*)d_in[7];

  float* out_f = (float*)d_out;
  float* cls_o   = out_f;              // [256, 256]
  float* alpha_o = out_f + 65536;      // [256, 8, 257, 257]

  // Workspace carve-up (256B aligned). Total ~143 MB.
  char* ws = (char*)d_ws;
  size_t off = 0;
  auto carve = [&](size_t bytes) -> char* {
    char* p = ws + off;
    off = (off + bytes + 255) & ~(size_t)255;
    return p;
  };
  const size_t padHalves = (size_t)B_ * LPAD * D_;      // 17,825,792
  _Float16* P    = (_Float16*)carve(padHalves * 2);
  _Float16* Qh   = (_Float16*)carve(padHalves * 2);
  _Float16* Kh   = (_Float16*)carve(padHalves * 2);
  _Float16* Vh   = (_Float16*)carve(padHalves * 2);
  _Float16* WqT  = (_Float16*)carve((size_t)D_ * D_ * 2);
  _Float16* WkT  = (_Float16*)carve((size_t)D_ * D_ * 2);
  _Float16* WvT  = (_Float16*)carve((size_t)D_ * D_ * 2);
  _Float16* WoT  = (_Float16*)carve((size_t)D_ * D_ * 2);
  _Float16* HV16 = (_Float16*)carve((size_t)B_ * D_ * 2);
  (void)ws_size; (void)in_sizes; (void)n_in; (void)out_size;

  prep_pad<<<B_, 256, 0, stream>>>(x, lengths, cls_emb, P);
  conv_w<<<256, 256, 0, stream>>>(Wq, Wk, Wv, Wo, WqT, WkT, WvT, WoT);
  qkv_gemm<<<dim3(MT, 3, B_), 256, 0, stream>>>(P, WqT, WkT, WvT, Qh, Kh, Vh);
  attn_softmax<<<dim3(MT, H_, B_), 256, 0, stream>>>(Qh, Kh, Vh, lengths, alpha_o, HV16);
  cls_out<<<32, 256, 0, stream>>>(HV16, WoT, cls_o);
}
